// MultiHeadAttention_53395033424620
// MI455X (gfx1250) — matrix operand order repaired and checked by reading, not compiled, not run on hardware
//
#include <hip/hip_runtime.h>
#include <hip/hip_bf16.h>

typedef __bf16 bf16;
typedef __attribute__((ext_vector_type(16))) __bf16 v16bf;
typedef __attribute__((ext_vector_type(8)))  __bf16 v8bf;
typedef __attribute__((ext_vector_type(8)))  float  v8f;
typedef __attribute__((ext_vector_type(4)))  int    v4i;

// Problem constants (fixed by the reference)
constexpr int BS = 2, T = 2048, D = 1024, H = 16, DH = 64;
constexpr int TOK = BS * T;        // 4096 rows
constexpr int LD3 = 3 * D;         // kqv row stride = 3072

#define TRY_ASYNC 1

// ---------------------------------------------------------------------------
// WMMA helper: D = A(16x32 bf16) x B(32x16 bf16) + C(16x16 f32)
// ---------------------------------------------------------------------------
__device__ inline v8f wmma_bf16(v16bf a, v16bf b, v8f c) {
  return __builtin_amdgcn_wmma_f32_16x16x32_bf16(
      /*neg_a=*/false, a, /*neg_b=*/false, b,
      /*c_mod=*/(short)0, c, /*reuse_a=*/false, /*reuse_b=*/false);
}

// A fragment (16x32, row-major source, leading dim ld).
__device__ inline v16bf frag_a(const bf16* __restrict__ base, int ld, int lane) {
  const bf16* p = base + (lane & 15) * ld;
  const int koff = (lane < 16) ? 0 : 16;
  v16bf a;
#pragma unroll
  for (int e = 0; e < 8; ++e) {
    a[e]     = p[koff + e];
    a[8 + e] = p[8 + koff + e];
  }
  return a;
}

// B fragment (32x16, row-major K x N source, leading dim ld).
__device__ inline v16bf frag_b(const bf16* __restrict__ base, int ld, int lane) {
  const int n  = lane & 15;
  const int kb = (lane < 16) ? 0 : 16;
  v16bf b;
#pragma unroll
  for (int j = 0; j < 8; ++j) {
    b[2 * j]     = base[(kb + 2 * j) * ld + n];
    b[2 * j + 1] = base[(kb + 2 * j + 1) * ld + n];
  }
  return b;
}

// B fragment from transposed storage: element (K,N) read from row-major [N][K]
// (contiguous along K -> vectorizes to b128). Used for Q@K^T and P@V (with Vt).
__device__ inline v16bf frag_b_trans(const bf16* __restrict__ base, int ld,
                                     int nbase, int kbase, int lane) {
  const bf16* p = base + (nbase + (lane & 15)) * ld + kbase + ((lane < 16) ? 0 : 16);
  v16bf b;
#pragma unroll
  for (int j = 0; j < 16; ++j) b[j] = p[j];
  return b;
}

// ---------------------------------------------------------------------------
// 16-byte global -> LDS copy (async-to-LDS when available)
// Builtin signature (from toolchain diagnostics):
//   (v4i32 addrspace(1)* global_src, v4i32 addrspace(3)* lds_dst, Ii off, Ii cpol)
// ---------------------------------------------------------------------------
__device__ inline void copy16(const bf16* __restrict__ g, bf16* l) {
#if TRY_ASYNC && __has_builtin(__builtin_amdgcn_global_load_async_to_lds_b128)
  __builtin_amdgcn_global_load_async_to_lds_b128(
      (__attribute__((address_space(1))) v4i*)const_cast<bf16*>(g),
      (__attribute__((address_space(3))) v4i*)l, 0, 0);
#else
  *(v8bf*)l = *(const v8bf*)g;
#endif
}

__device__ inline void wait_async0() {
#if TRY_ASYNC && __has_builtin(__builtin_amdgcn_global_load_async_to_lds_b128)
#if __has_builtin(__builtin_amdgcn_s_wait_asynccnt)
  __builtin_amdgcn_s_wait_asynccnt(0);
#else
  asm volatile("s_wait_asynccnt 0" ::: "memory");
#endif
#endif
}

// ---------------------------------------------------------------------------
// fp32 -> bf16 cast
// ---------------------------------------------------------------------------
__global__ __launch_bounds__(256) void cast_f32_bf16(const float* __restrict__ in,
                                                     bf16* __restrict__ out, int n) {
  int i = blockIdx.x * 256 + threadIdx.x;
  if (i < n) out[i] = (bf16)in[i];
}

// ---------------------------------------------------------------------------
// V transpose: kqv v-chunk [b, t, h*64+d] -> vt [b, h, d, t]  (64x64 LDS tiles)
// ---------------------------------------------------------------------------
__global__ __launch_bounds__(256) void transpose_v(const bf16* __restrict__ kqv,
                                                   bf16* __restrict__ vt) {
  constexpr int LDT = 72;
  __shared__ bf16 tile[64 * LDT];
  const int tid = threadIdx.x;
  const int t0 = blockIdx.x * 64, h = blockIdx.y, b = blockIdx.z;
  const bf16* src = kqv + (size_t)b * T * LD3 + 2 * D + h * DH;  // [t][d], ld=3072

  // load 64(t) x 64(d): 8-elem chunks, 2 per thread, coalesced along d
  {
    const int tl = tid >> 3, c = (tid & 7) * 8;
#pragma unroll
    for (int p = 0; p < 64; p += 32)
      *(v8bf*)&tile[(tl + p) * LDT + c] =
          *(const v8bf*)&src[(size_t)(t0 + tl + p) * LD3 + c];
  }
  __syncthreads();
  // store 64(d) x 64(t): contiguous along t
  bf16* dst = vt + (size_t)(b * H + h) * DH * T;
  {
    const int dl = tid >> 3, tc = (tid & 7) * 8;
#pragma unroll
    for (int p = 0; p < 64; p += 32) {
      v8bf v;
#pragma unroll
      for (int e = 0; e < 8; ++e) v[e] = tile[(tc + e) * LDT + dl + p];
      *(v8bf*)&dst[(size_t)(dl + p) * T + t0 + tc] = v;
    }
  }
}

// ---------------------------------------------------------------------------
// LDS-staged WMMA GEMM with bias: C[M,N] = A[M,K] @ B[K,N] + bias[N]
// Block tile 128(M) x 64(N), K-step 32. 8 waves: 4(M) x 2(N), 32x32 per wave.
// A/B tiles staged in LDS via (async) 16B copies, shared by all 8 waves.
// ---------------------------------------------------------------------------
template <bool F32OUT>
__global__ __launch_bounds__(256) void gemm_bias(
    const bf16* __restrict__ A, const bf16* __restrict__ B,
    const float* __restrict__ bias, float* __restrict__ Cf,
    bf16* __restrict__ Cb, int M, int N, int K) {
  constexpr int LDA = 40;   // padded strides (bf16 units, 16B-aligned)
  constexpr int LDB = 72;
  __shared__ bf16 Abuf[128 * LDA];
  __shared__ bf16 Bbuf[32 * LDB];

  const int tid  = threadIdx.x;
  const int lane = tid & 31;
  const int wave = tid >> 5;
  const int m0b = blockIdx.y * 128, n0b = blockIdx.x * 64;
  const int wm = wave >> 1, wn = wave & 1;

  // staging coordinates (one 16B chunk each; A needs 2 passes)
  const int arow = tid >> 2, ac = (tid & 3) * 8;   // 64 rows/pass x 4 chunks
  const int brow = tid >> 3, bc = (tid & 7) * 8;   // 32 rows x 8 chunks

  v8f acc00 = {}, acc01 = {}, acc10 = {}, acc11 = {};
  for (int k0 = 0; k0 < K; k0 += 32) {
    __syncthreads();   // previous iteration's reads done before overwrite
    copy16(A + (size_t)(m0b + arow) * K + k0 + ac, Abuf + arow * LDA + ac);
    copy16(A + (size_t)(m0b + arow + 64) * K + k0 + ac, Abuf + (arow + 64) * LDA + ac);
    copy16(B + (size_t)(k0 + brow) * N + n0b + bc, Bbuf + brow * LDB + bc);
    wait_async0();
    __syncthreads();

    v16bf a0 = frag_a(Abuf + (wm * 32) * LDA, LDA, lane);
    v16bf a1 = frag_a(Abuf + (wm * 32 + 16) * LDA, LDA, lane);
    v16bf b0 = frag_b(Bbuf + wn * 32, LDB, lane);
    v16bf b1 = frag_b(Bbuf + wn * 32 + 16, LDB, lane);
    acc00 = wmma_bf16(a0, b0, acc00);
    acc01 = wmma_bf16(a0, b1, acc01);
    acc10 = wmma_bf16(a1, b0, acc10);
    acc11 = wmma_bf16(a1, b1, acc11);
  }

  const int m0 = m0b + wm * 32, n0 = n0b + wn * 32;
  const int col = lane & 15;
  const int rb  = (lane < 16) ? 0 : 8;
#pragma unroll
  for (int r = 0; r < 8; ++r) {
    const int row0 = m0 + rb + r;
    const int row1 = m0 + 16 + rb + r;
    const int c0 = n0 + col, c1 = n0 + 16 + col;
    float v00 = acc00[r] + bias[c0];
    float v01 = acc01[r] + bias[c1];
    float v10 = acc10[r] + bias[c0];
    float v11 = acc11[r] + bias[c1];
    if constexpr (F32OUT) {
      Cf[(size_t)row0 * N + c0] = v00; Cf[(size_t)row0 * N + c1] = v01;
      Cf[(size_t)row1 * N + c0] = v10; Cf[(size_t)row1 * N + c1] = v11;
    } else {
      Cb[(size_t)row0 * N + c0] = (bf16)v00; Cb[(size_t)row0 * N + c1] = (bf16)v01;
      Cb[(size_t)row1 * N + c0] = (bf16)v10; Cb[(size_t)row1 * N + c1] = (bf16)v11;
    }
  }
}

// ---------------------------------------------------------------------------
// Flash attention (causal). kqv layout [b, t, 3D] chunk order k,q,v; V taken
// from vt [b, h, d, t] so every fragment load is contiguous (b128).
// One block per (128-query chunk, head, batch); each of 8 waves owns 16 query
// rows and streams 32-key blocks with fp32 online softmax.
// ---------------------------------------------------------------------------
__global__ __launch_bounds__(256) void flash_attn(const bf16* __restrict__ kqv,
                                                  const bf16* __restrict__ vt,
                                                  bf16* __restrict__ att) {
  constexpr int LDP = 40;
  __shared__ bf16 pbuf[8 * 16 * LDP];

  const int lane = threadIdx.x & 31;
  const int wave = threadIdx.x >> 5;
  const int qblk = blockIdx.x;
  const int h    = blockIdx.y;
  const int b    = blockIdx.z;
  const int q0   = qblk * 128 + wave * 16;

  const bf16* base = kqv + (size_t)b * T * LD3;
  const bf16* Kp = base + h * DH;            // k chunk (offset 0), [t][d]
  const bf16* Qp = base + D + h * DH;        // q chunk
  const bf16* Vt = vt + (size_t)(b * H + h) * DH * T;   // [d][t]

  const v16bf qf0 = frag_a(Qp + (size_t)q0 * LD3 + 0, LD3, lane);
  const v16bf qf1 = frag_a(Qp + (size_t)q0 * LD3 + 32, LD3, lane);

  v8f acc[4] = {{}, {}, {}, {}};
  float mrow[8], lrow[8];
#pragma unroll
  for (int r = 0; r < 8; ++r) { mrow[r] = -__builtin_inff(); lrow[r] = 0.f; }

  const int col = lane & 15;
  const int rb  = (lane < 16) ? 0 : 8;
  bf16* pw = pbuf + wave * 16 * LDP;
  const float scale = 0.125f;                // 1/sqrt(64)

  for (int j0 = 0; j0 < q0 + 16; j0 += 32) {
    // ---- S = Q @ K^T for 32 keys (two 16x16 tiles) ----
    v8f z = {};
    v8f s0 = wmma_bf16(qf0, frag_b_trans(Kp, LD3, j0, 0, lane), z);
    s0     = wmma_bf16(qf1, frag_b_trans(Kp, LD3, j0, 32, lane), s0);
    v8f s1 = wmma_bf16(qf0, frag_b_trans(Kp, LD3, j0 + 16, 0, lane), z);
    s1     = wmma_bf16(qf1, frag_b_trans(Kp, LD3, j0 + 16, 32, lane), s1);

    // ---- online softmax (fp32) ----
    float p0a[8], p1a[8];
#pragma unroll
    for (int r = 0; r < 8; ++r) {
      const int grow = q0 + rb + r;
      const int key0 = j0 + col;
      float e0 = (key0 <= grow) ? s0[r] * scale : -__builtin_inff();
      float e1 = (key0 + 16 <= grow) ? s1[r] * scale : -__builtin_inff();
      float v = fmaxf(e0, e1);
#pragma unroll
      for (int off = 1; off < 16; off <<= 1) v = fmaxf(v, __shfl_xor(v, off, 32));
      const float mnew  = fmaxf(mrow[r], v);
      const float alpha = __expf(mrow[r] - mnew);
      mrow[r] = mnew;
      const float p0 = __expf(e0 - mnew);
      const float p1 = __expf(e1 - mnew);
      float s = p0 + p1;
#pragma unroll
      for (int off = 1; off < 16; off <<= 1) s += __shfl_xor(s, off, 32);
      lrow[r] = lrow[r] * alpha + s;
#pragma unroll
      for (int n = 0; n < 4; ++n) acc[n][r] *= alpha;
      p0a[r] = p0; p1a[r] = p1;
    }

    // ---- transpose P (C-layout fp32) -> A-layout bf16 via per-wave LDS ----
#pragma unroll
    for (int r = 0; r < 8; ++r) {
      pw[(rb + r) * LDP + col]      = (bf16)p0a[r];
      pw[(rb + r) * LDP + 16 + col] = (bf16)p1a[r];
    }
    asm volatile("s_wait_dscnt 0" ::: "memory");   // same-wave DS RAW ordering
    v16bf pf;
    {
      const bf16* pr = pw + (lane & 15) * LDP;
      const int koff = (lane < 16) ? 0 : 16;
#pragma unroll
      for (int e = 0; e < 8; ++e) {
        pf[e]     = pr[koff + e];
        pf[8 + e] = pr[8 + koff + e];
      }
    }
    asm volatile("" ::: "memory");

    // ---- acc += P(16x32) @ V(32x64), V fragments from [d][t] storage ----
#pragma unroll
    for (int n = 0; n < 4; ++n)
      acc[n] = wmma_bf16(pf, frag_b_trans(Vt, T, n * 16, j0, lane), acc[n]);
  }

  // ---- normalize and store: att[b, t, h*64 + d] ----
  bf16* outp = att + (size_t)b * T * D + h * DH;
#pragma unroll
  for (int r = 0; r < 8; ++r) {
    const float inv = 1.f / lrow[r];
    const int row = q0 + rb + r;
#pragma unroll
    for (int n = 0; n < 4; ++n)
      outp[(size_t)row * D + n * 16 + col] = (bf16)(acc[n][r] * inv);
  }
}

// ---------------------------------------------------------------------------
// Host launcher
// ---------------------------------------------------------------------------
extern "C" void kernel_launch(void* const* d_in, const int* in_sizes, int n_in,
                              void* d_out, int out_size, void* d_ws, size_t ws_size,
                              hipStream_t stream) {
  const float* x  = (const float*)d_in[0];   // [BS, T, D]
  const float* W1 = (const float*)d_in[1];   // [D, 3D]
  const float* b1 = (const float*)d_in[2];   // [3D]
  const float* W2 = (const float*)d_in[3];   // [D, D]
  const float* b2 = (const float*)d_in[4];   // [D]
  float* out = (float*)d_out;                // [BS, T, D] fp32

  const int xN   = TOK * D;
  const int w1N  = D * 3 * D;
  const int w2N  = D * D;
  const int kqvN = TOK * 3 * D;
  const int attN = TOK * D;
  const int vtN  = BS * H * DH * T;

  bf16* xb   = (bf16*)d_ws;
  bf16* w1b  = xb + xN;
  bf16* w2b  = w1b + w1N;
  bf16* kqvb = w2b + w2N;
  bf16* attb = kqvb + kqvN;
  bf16* vtb  = attb + attN;
  (void)in_sizes; (void)n_in; (void)out_size; (void)ws_size; (void)vtN;

  // 1) casts to bf16
  cast_f32_bf16<<<(xN + 255) / 256, 256, 0, stream>>>(x, xb, xN);
  cast_f32_bf16<<<(w1N + 255) / 256, 256, 0, stream>>>(W1, w1b, w1N);
  cast_f32_bf16<<<(w2N + 255) / 256, 256, 0, stream>>>(W2, w2b, w2N);

  // 2) kqv = x @ W1 + b1   (bf16 out, fp32 accumulate)
  gemm_bias<false><<<dim3(3 * D / 64, TOK / 128), 256, 0, stream>>>(
      xb, w1b, b1, nullptr, kqvb, TOK, 3 * D, D);

  // 3) V -> [b, h, d, t] for contiguous P@V fragments
  transpose_v<<<dim3(T / 64, H, BS), 256, 0, stream>>>(kqvb, vtb);

  // 4) causal flash attention
  flash_attn<<<dim3(T / 128, H, BS), 256, 0, stream>>>(kqvb, vtb, attb);

  // 5) out = att @ W2 + b2  (fp32 out)
  gemm_bias<true><<<dim3(D / 64, TOK / 128), 256, 0, stream>>>(
      attb, w2b, b2, out, nullptr, TOK, D, D);
}
